// SirenNeRFModel_57784490000774
// MI455X (gfx1250) — matrix operand order repaired and checked by reading, not compiled, not run on hardware
//
#include <hip/hip_runtime.h>
#include <math.h>

typedef __attribute__((ext_vector_type(16))) _Float16 v16h;
typedef __attribute__((ext_vector_type(8)))  float    v8f;
typedef __attribute__((ext_vector_type(4)))  float    v4f;

#define NOPS 40
#define WLDS_HALVES (NOPS * 512)   // per operand: 32 lanes * 16 halves

// omega0 / (2*pi): v_sin_f32 computes sin(2*pi*x)
#define W0_OVER_2PI 4.774648292756860f

union V16 { v16h h; unsigned int u[8]; };

// Opaque pass-through: prevents the compiler from hoisting/CSE'ing the LDS
// loads of the weight operands out of the tile loop (which caused scratch
// spills of ~320 VGPRs of hoisted weights in the previous revision).
__device__ __forceinline__ unsigned opaque_u(unsigned x) {
  asm volatile("" : "+v"(x));
  return x;
}

__device__ __forceinline__ v8f zero8() {
  v8f z;
#pragma unroll
  for (int i = 0; i < 8; ++i) z[i] = 0.f;
  return z;
}

__device__ __forceinline__ unsigned pk2(float x, float y) {
  _Float16 hx = (_Float16)x, hy = (_Float16)y;
  unsigned short ux = __builtin_bit_cast(unsigned short, hx);
  unsigned short uy = __builtin_bit_cast(unsigned short, hy);
  return (unsigned)ux | ((unsigned)uy << 16);
}

// swap lanes 0-15 <-> 16-31 (ds_swizzle BITMASK: and=0x1f, or=0, xor=0x10)
__device__ __forceinline__ unsigned swap16u(unsigned x) {
  return (unsigned)__builtin_amdgcn_ds_swizzle((int)x, 0x401F);
}
__device__ __forceinline__ float swap16f(float x) {
  return __uint_as_float(swap16u(__float_as_uint(x)));
}

__device__ __forceinline__ v8f wmma_acc(const unsigned short* w, int op, int lane,
                                        v16h b, v8f c) {
  const v16h* ap =
      reinterpret_cast<const v16h*>(w + opaque_u((unsigned)(op * 512 + lane * 16)));
  v16h a = *ap;
  return __builtin_amdgcn_wmma_f32_16x16x32_f16(false, a, false, b, (short)0, c,
                                                false, false);
}

// Dense 64-out sine layer: 4 output tiles, nch K-chunks of 32.
__device__ __forceinline__ void sineLayer64(const unsigned short* w, const float* bl,
                                            int opbase, int bbase, int nch,
                                            int lane, int half16,
                                            const v16h bin[2], float s[4][8]) {
#pragma unroll
  for (int t = 0; t < 4; ++t) {
    v8f c = zero8();
#pragma unroll
    for (int ci = 0; ci < 2; ++ci)
      if (ci < nch) c = wmma_acc(w, opbase + t * nch + ci, lane, bin[ci], c);
    unsigned bb = opaque_u((unsigned)(bbase + t * 16 + half16 * 8));
    v4f b0 = *reinterpret_cast<const v4f*>(bl + bb);
    v4f b1 = *reinterpret_cast<const v4f*>(bl + bb + 4);
#pragma unroll
    for (int r = 0; r < 8; ++r) {
      float bias = (r < 4) ? b0[r] : b1[r - 4];
      s[t][r] = __builtin_amdgcn_sinf(W0_OVER_2PI * (c[r] + bias));
    }
  }
}

// Convert D-layout activations (s[tile][vgpr]) into next layer's B operands.
__device__ __forceinline__ void packB(const float s[4][8], int half16, v16h bout[2]) {
#pragma unroll
  for (int cc = 0; cc < 2; ++cc) {
    V16 b;
    const int tlo = cc * 2, thi = cc * 2 + 1;
#pragma unroll
    for (int v = 0; v < 4; ++v) {
      unsigned lo = pk2(s[tlo][2 * v], s[tlo][2 * v + 1]);
      unsigned hi = pk2(s[thi][2 * v], s[thi][2 * v + 1]);
      unsigned losw = swap16u(lo);
      unsigned hisw = swap16u(hi);
      b.u[v]     = half16 ? hisw : lo;
      b.u[v + 4] = half16 ? hi   : losw;
    }
    bout[cc] = b.h;
  }
}

// Positional-encoding value for concat-dim K of one point (K>=60 -> pad 0).
// sin(2^i * pi * x) == sin_hw(x * 2^(i-1)) where sin_hw(t) = sin(2*pi*t).
__device__ __forceinline__ float peval(float pn0, float pn1, float pn2, int K) {
  float r = 0.f;
  if (K < 60) {
    int coord = (K >= 40) ? 2 : ((K >= 20) ? 1 : 0);
    int j = K - coord * 20;
    int i = (j < 10) ? j : j - 10;
    float pv = (coord == 0) ? pn0 : ((coord == 1) ? pn1 : pn2);
    float rev = pv * (0.5f * (float)(1 << i));
    r = (j < 10) ? __builtin_amdgcn_sinf(rev) : __builtin_amdgcn_cosf(rev);
  }
  return r;
}

__global__ __launch_bounds__(256) void siren_nerf_kernel(
    const float* __restrict__ pts, const float* __restrict__ views,
    const float* __restrict__ aabb,
    const float* __restrict__ W1, const float* __restrict__ b1,
    const float* __restrict__ W2, const float* __restrict__ b2,
    const float* __restrict__ W3, const float* __restrict__ b3,
    const float* __restrict__ W4, const float* __restrict__ b4,
    const float* __restrict__ W5, const float* __restrict__ b5,
    const float* __restrict__ W6, const float* __restrict__ b6,
    const float* __restrict__ W7, const float* __restrict__ b7,
    float* __restrict__ colors, float* __restrict__ dens, int N, int nTiles) {
  __shared__ __align__(32) unsigned short wlds[WLDS_HALVES];
  __shared__ __align__(32) float blds[352];

  const int tid = threadIdx.x;

  // ---- preload weights into LDS as f16 in WMMA A-operand layout (A = W^T) ----
  for (int e = tid; e < WLDS_HALVES; e += blockDim.x) {
    int o = e >> 9;
    int rem = e & 511;
    int l = rem >> 4;
    int h = rem & 15;
    int L = (o >= 8) + (o >= 16) + (o >= 24) + (o >= 26) + (o >= 30) + (o >= 38);
    int base, ch, Kd, Nd;
    const float* W;
    switch (L) {
      case 0: base = 0;  ch = 2; Kd = 60; Nd = 64; W = W1; break;
      case 1: base = 8;  ch = 2; Kd = 64; Nd = 64; W = W2; break;
      case 2: base = 16; ch = 2; Kd = 64; Nd = 64; W = W3; break;
      case 3: base = 24; ch = 2; Kd = 64; Nd = 16; W = W4; break;
      case 4: base = 26; ch = 1; Kd = 31; Nd = 64; W = W5; break;
      case 5: base = 30; ch = 2; Kd = 64; Nd = 64; W = W6; break;
      default:base = 38; ch = 2; Kd = 64; Nd = 3;  W = W7; break;
    }
    int local = o - base;
    int t = local / ch;
    int c = local - t * ch;
    // A order used here (contiguous K, same as B): lane<16: K = h; lanes 16-31: K = h + 16
    int Klocal = h + ((l & 16) ? 16 : 0);
    int k = c * 32 + Klocal;
    int m = t * 16 + (l & 15);
    float val = (k < Kd && m < Nd) ? W[k * Nd + m] : 0.f;
    _Float16 hv = (_Float16)val;
    wlds[e] = __builtin_bit_cast(unsigned short, hv);
  }
  // ---- preload biases (padded) ----
  for (int e = tid; e < 352; e += blockDim.x) {
    int L = (e >= 64) + (e >= 128) + (e >= 192) + (e >= 208) + (e >= 272) + (e >= 336);
    const float* B;
    int base, act;
    switch (L) {
      case 0: B = b1; base = 0;   act = 64; break;
      case 1: B = b2; base = 64;  act = 64; break;
      case 2: B = b3; base = 128; act = 64; break;
      case 3: B = b4; base = 192; act = 16; break;
      case 4: B = b5; base = 208; act = 64; break;
      case 5: B = b6; base = 272; act = 64; break;
      default:B = b7; base = 336; act = 3;  break;
    }
    int off = e - base;
    blds[e] = (off < act) ? B[off] : 0.f;
  }
  __syncthreads();

  const int lane = tid & 31;
  const int half16 = lane >> 4;
  const int waveId = blockIdx.x * (blockDim.x >> 5) + (tid >> 5);
  const int nWaves = gridDim.x * (blockDim.x >> 5);

  const float a0x = aabb[0], a0y = aabb[1], a0z = aabb[2];
  const float r0 = 1.f / (aabb[3] - a0x);
  const float r1 = 1.f / (aabb[4] - a0y);
  const float r2 = 1.f / (aabb[5] - a0z);

  for (int tile = waveId; tile < nTiles; tile += nWaves) {
    const int p = tile * 16 + (lane & 15);
    const int pc = (p < N) ? p : (N - 1);

    const float pn0 = (pts[pc * 3 + 0] - a0x) * r0;
    const float pn1 = (pts[pc * 3 + 1] - a0y) * r1;
    const float pn2 = (pts[pc * 3 + 2] - a0z) * r2;

    // ---- positional encoding directly in B-operand layout (2 chunks of K=32) ----
    v16h bin[2];
#pragma unroll
    for (int cc = 0; cc < 2; ++cc) {
      V16 b;
#pragma unroll
      for (int v = 0; v < 8; ++v) {
        int K0 = cc * 32 + half16 * 16 + 2 * v;
        b.u[v] = pk2(peval(pn0, pn1, pn2, K0), peval(pn0, pn1, pn2, K0 + 1));
      }
      bin[cc] = b.h;
    }

    float s[4][8];
    // L1: 60(pad64) -> 64, L2/L3: 64 -> 64
    sineLayer64(wlds, blds, 0, 0, 2, lane, half16, bin, s);
    packB(s, half16, bin);
    sineLayer64(wlds, blds, 8, 64, 2, lane, half16, bin, s);
    packB(s, half16, bin);
    sineLayer64(wlds, blds, 16, 128, 2, lane, half16, bin, s);
    packB(s, half16, bin);

    // L4: 64 -> 16 (linear)
    v8f c4 = zero8();
    c4 = wmma_acc(wlds, 24, lane, bin[0], c4);
    c4 = wmma_acc(wlds, 25, lane, bin[1], c4);
    float x4[8];
    {
      unsigned bb = opaque_u((unsigned)(192 + half16 * 8));
      v4f b0 = *reinterpret_cast<const v4f*>(blds + bb);
      v4f b1v = *reinterpret_cast<const v4f*>(blds + bb + 4);
#pragma unroll
      for (int r = 0; r < 8; ++r) x4[r] = c4[r] + ((r < 4) ? b0[r] : b1v[r - 4]);
    }
    if (half16 == 0 && p < N) dens[p] = fmaxf(x4[0], 0.f);
    float x4sw[8];
#pragma unroll
    for (int r = 0; r < 8; ++r) x4sw[r] = swap16f(x4[r]);

    // ---- spherical harmonics of normalized view dir ----
    float vx = views[pc * 3 + 0], vy = views[pc * 3 + 1], vz = views[pc * 3 + 2];
    float inv = rsqrtf(vx * vx + vy * vy + vz * vz);
    vx *= inv; vy *= inv; vz *= inv;
    const float xx = vx * vx, yy = vy * vy, zz = vz * vz;
    const float xy = vx * vy, yz = vy * vz, xz = vx * vz;
    float sh[16];
    sh[0]  = 0.28209479177387814f;
    sh[1]  = -0.4886025119029199f * vy;
    sh[2]  =  0.4886025119029199f * vz;
    sh[3]  = -0.4886025119029199f * vx;
    sh[4]  =  1.0925484305920792f * xy;
    sh[5]  = -1.0925484305920792f * yz;
    sh[6]  =  0.31539156525252005f * (3.f * zz - 1.f);
    sh[7]  = -1.0925484305920792f * xz;
    sh[8]  =  0.5462742152960396f * (xx - yy);
    sh[9]  = -0.5900435899266435f * vy * (3.f * xx - yy);
    sh[10] =  2.890611442640554f * xy * vz;
    sh[11] = -0.4570457994644658f * vy * (5.f * zz - 1.f);
    sh[12] =  0.3731763325901154f * vz * (5.f * zz - 3.f);
    sh[13] = -0.4570457994644658f * vx * (5.f * zz - 1.f);
    sh[14] =  1.445305721320277f * vz * (xx - yy);
    sh[15] = -0.5900435899266435f * vx * (xx - 3.f * yy);

    // ---- build layer-5 input B: concat(x4[1:16], sh16), K padded to 32 ----
    float val[16];
#pragma unroll
    for (int k = 0; k < 16; ++k) {
      float vlow  = (k < 7) ? x4[k + 1] : ((k < 15) ? x4sw[k - 7] : sh[0]);
      float vhigh = (k < 15) ? sh[k + 1] : 0.f;
      val[k] = half16 ? vhigh : vlow;
    }
    V16 b5u;
#pragma unroll
    for (int v = 0; v < 8; ++v) b5u.u[v] = pk2(val[2 * v], val[2 * v + 1]);
    bin[0] = b5u.h;

    // L5: 31(pad32) -> 64, L6: 64 -> 64
    sineLayer64(wlds, blds, 26, 208, 1, lane, half16, bin, s);
    packB(s, half16, bin);
    sineLayer64(wlds, blds, 30, 272, 2, lane, half16, bin, s);
    packB(s, half16, bin);

    // L7: 64 -> 3(pad16), sigmoid
    v8f c7 = zero8();
    c7 = wmma_acc(wlds, 38, lane, bin[0], c7);
    c7 = wmma_acc(wlds, 39, lane, bin[1], c7);
    if (half16 == 0 && p < N) {
#pragma unroll
      for (int j = 0; j < 3; ++j) {
        float x = c7[j] + blds[336 + j];
        colors[p * 3 + j] = 1.f / (1.f + __expf(-x));
      }
    }
  }
}

extern "C" void kernel_launch(void* const* d_in, const int* in_sizes, int n_in,
                              void* d_out, int out_size, void* d_ws, size_t ws_size,
                              hipStream_t stream) {
  const float* pts   = (const float*)d_in[0];
  const float* views = (const float*)d_in[1];
  const float* aabb  = (const float*)d_in[2];
  const float* W1 = (const float*)d_in[3];
  const float* b1 = (const float*)d_in[4];
  const float* W2 = (const float*)d_in[5];
  const float* b2 = (const float*)d_in[6];
  const float* W3 = (const float*)d_in[7];
  const float* b3 = (const float*)d_in[8];
  const float* W4 = (const float*)d_in[9];
  const float* b4 = (const float*)d_in[10];
  const float* W5 = (const float*)d_in[11];
  const float* b5 = (const float*)d_in[12];
  const float* W6 = (const float*)d_in[13];
  const float* b6 = (const float*)d_in[14];
  const float* W7 = (const float*)d_in[15];
  const float* b7 = (const float*)d_in[16];

  const int N = in_sizes[0] / 3;
  float* colors = (float*)d_out;
  float* dens   = colors + (size_t)3 * N;
  const int nTiles = (N + 15) / 16;

  const int threads = 256;
  const int wpb = threads / 32;
  long needed = ((long)nTiles + wpb - 1) / wpb;
  int blocks = (int)((needed < 2048) ? needed : 2048);
  if (blocks < 1) blocks = 1;

  hipLaunchKernelGGL(siren_nerf_kernel, dim3(blocks), dim3(threads), 0, stream,
                     pts, views, aabb, W1, b1, W2, b2, W3, b3, W4, b4, W5, b5,
                     W6, b6, W7, b7, colors, dens, N, nTiles);
}
